// RCCA_83829171683534
// MI455X (gfx1250) — matrix operand order repaired and checked by reading, not compiled, not run on hardware
//
#include <hip/hip_runtime.h>

// ---------------------------------------------------------------------------
// CDNA5 (gfx1250) criss-cross attention, R=2.
// All heavy einsums via v_wmma_f32_16x16x32_bf16 (bf16 A/B, f32 accum).
// wave32; fragments loaded directly from global, contiguous k = 16*(lane/16) + s.
// ---------------------------------------------------------------------------

typedef __bf16 bf16;
typedef __attribute__((ext_vector_type(8)))  bf16  bf16x8;   // 16 bytes
typedef __attribute__((ext_vector_type(16))) bf16  v16bf;    // 32 bytes
typedef __attribute__((ext_vector_type(8)))  float v8f;

#define BD   8
#define CH   512
#define CI   64
#define HH   97
#define HW   9409L          // 97*97
#define KP   128            // padded attention K (97 -> 128)
#define E2   256            // score row: [eH 0..127 | eW 128..255]
#define NEGI (-1000000000.0f)

union FragU { v16bf v; bf16x8 h[2]; };

// A fragment, 16x32 bf16 (MxK). rowbase already includes this lane's row.
// lanes 0-15 -> K 0..15; lanes 16-31 -> K 16..31 (matches B).
__device__ __forceinline__ v16bf load_frag_a(const bf16* rowbase, int k0, int lane) {
    const int kb = (lane < 16) ? 0 : 16;
    FragU f;
    f.h[0] = *reinterpret_cast<const bf16x8*>(rowbase + k0 + kb);
    f.h[1] = *reinterpret_cast<const bf16x8*>(rowbase + k0 + kb + 8);
    return f.v;
}

// B fragment, 32x16 bf16 (KxN). colbase already includes this lane's column
// (n = lane&15), stored K-contiguous. lanes 0-15: K 0..15, lanes 16-31: K 16..31.
__device__ __forceinline__ v16bf load_frag_b(const bf16* colbase, int k0, int lane) {
    const int ks = (lane < 16) ? 0 : 16;
    return *reinterpret_cast<const v16bf*>(colbase + k0 + ks);
}

__device__ __forceinline__ v8f wmma_bf16(v16bf a, v16bf b, v8f c) {
    return __builtin_amdgcn_wmma_f32_16x16x32_bf16(false, a, false, b, (short)0, c, false, false);
}

// ---------------------------------------------------------------------------
// Small utility kernels
// ---------------------------------------------------------------------------
__global__ void cvt_f32_bf16(const float* __restrict__ s, bf16* __restrict__ d, int n) {
    int i = blockIdx.x * 256 + threadIdx.x;
    if (i < n) d[i] = (bf16)s[i];
}

// zero the K-pad (g in [97,128)) of vT and vW once; valid region is rewritten
// every pass, the pad stays zero.
__global__ void pad_zero(bf16* __restrict__ vT, bf16* __restrict__ vW) {
    long idx = (long)blockIdx.x * 256 + threadIdx.x;
    long row = idx >> 5;
    int  g   = HH + (int)(idx & 31);
    if (row < (long)BD * CH * HH && g < KP) {
        vT[row * KP + g] = (bf16)0.0f;
        vW[row * KP + g] = (bf16)0.0f;
    }
}

// x (B,C,HW) f32 -> xT (B,HW,C) bf16  (coalesced read, scattered write)
__global__ void pack_xT(const float* __restrict__ x, bf16* __restrict__ xT) {
    long i = (long)blockIdx.x * 256 + threadIdx.x;
    if (i >= (long)BD * CH * HW) return;
    int  b = (int)(i / (CH * HW));
    long r = i % (CH * HW);
    int  c = (int)(r / HW);
    int  n = (int)(r % HW);
    xT[(b * HW + n) * CH + c] = (bf16)x[i];
}

// ---------------------------------------------------------------------------
// Projection GEMM (q or k): D(64 x HW) = W(64x512) * X(512 x HW) + bias
// A = W row-major (K contig), B = xT column-contiguous. Writes qT/kT (B,HW,64).
// grid: (148, B), block 256 (8 waves).
// ---------------------------------------------------------------------------
__global__ void gemm_qk(const bf16* __restrict__ Wt, const float* __restrict__ bias,
                        const bf16* __restrict__ xT, bf16* __restrict__ outT) {
    const int b    = blockIdx.y;
    const int n0   = blockIdx.x * 64;
    const int wv   = threadIdx.x >> 5, lane = threadIdx.x & 31;
    const int msub = wv & 3, ngrp = wv >> 2;

    const bf16* arow = Wt + (msub * 16 + (lane & 15)) * CH;
    const bf16* xTb  = xT + (long)b * HW * CH;

    const bf16* bcol[2];
    int ncol[2];
    #pragma unroll
    for (int t = 0; t < 2; ++t) {
        int n   = n0 + (ngrp + 2 * t) * 16 + (lane & 15);
        ncol[t] = n;
        bcol[t] = xTb + (long)(n < (int)HW ? n : (int)HW - 1) * CH;
    }

    v8f acc[2] = {};
    for (int k0 = 0; k0 < CH; k0 += 32) {
        v16bf a = load_frag_a(arow, k0, lane);
        __builtin_prefetch(bcol[0] + k0 + 64, 0, 0);
        #pragma unroll
        for (int t = 0; t < 2; ++t) {
            v16bf bb = load_frag_b(bcol[t], k0, lane);
            acc[t] = wmma_bf16(a, bb, acc[t]);
        }
    }

    const int moff = msub * 16 + ((lane < 16) ? 0 : 8);
    #pragma unroll
    for (int t = 0; t < 2; ++t) {
        if (ncol[t] < (int)HW) {
            bf16x8 pk;
            #pragma unroll
            for (int r = 0; r < 8; ++r) pk[r] = (bf16)(acc[t][r] + bias[moff + r]);
            *reinterpret_cast<bf16x8*>(outT + ((long)b * HW + ncol[t]) * CI + moff) = pk;
        }
    }
}

// ---------------------------------------------------------------------------
// v projection: D(512 x HW) = Wv * X + bv ; scatter to vT (B,C,W,128g) and
// vW (B,C,H,128w).  grid: (148, 8, B).
// ---------------------------------------------------------------------------
__global__ void gemm_v(const bf16* __restrict__ Wv, const float* __restrict__ bias,
                       const bf16* __restrict__ xT,
                       bf16* __restrict__ vT, bf16* __restrict__ vW) {
    const int b = blockIdx.z, mt = blockIdx.y;
    const int n0 = blockIdx.x * 64;
    const int wv = threadIdx.x >> 5, lane = threadIdx.x & 31;
    const int msub = wv & 3, ngrp = wv >> 2;

    const bf16* arow = Wv + (long)(mt * 64 + msub * 16 + (lane & 15)) * CH;
    const bf16* xTb  = xT + (long)b * HW * CH;

    const bf16* bcol[2];
    int ncol[2];
    #pragma unroll
    for (int t = 0; t < 2; ++t) {
        int n   = n0 + (ngrp + 2 * t) * 16 + (lane & 15);
        ncol[t] = n;
        bcol[t] = xTb + (long)(n < (int)HW ? n : (int)HW - 1) * CH;
    }

    v8f acc[2] = {};
    for (int k0 = 0; k0 < CH; k0 += 32) {
        v16bf a = load_frag_a(arow, k0, lane);
        #pragma unroll
        for (int t = 0; t < 2; ++t) {
            v16bf bb = load_frag_b(bcol[t], k0, lane);
            acc[t] = wmma_bf16(a, bb, acc[t]);
        }
    }

    const int cbase = mt * 64 + msub * 16 + ((lane < 16) ? 0 : 8);
    #pragma unroll
    for (int t = 0; t < 2; ++t) {
        int n = ncol[t];
        if (n < (int)HW) {
            int g = n / HH, w = n % HH;
            #pragma unroll
            for (int r = 0; r < 8; ++r) {
                int c = cbase + r;
                float val = acc[t][r] + bias[c];
                vT[(((long)b * CH + c) * HH + w) * KP + g] = (bf16)val;
                vW[(((long)b * CH + c) * HH + g) * KP + w] = (bf16)val;
            }
        }
    }
}

// ---------------------------------------------------------------------------
// Score GEMMs. mode 0: eH per (b,w):  e[m=h][n=g] = sum_c qT[h,w]*kT[g,w]
//             mode 1: eW per (b,h):  e[m=w][n=v] = sum_c qT[h,w]*kT[h,v]
// 97 padded to 112: 7x7 = 49 subtiles round-robined over 8 waves. K=64.
// Output escore (B,HW,256) f32 at column offset mode*128. grid: (97, B).
// ---------------------------------------------------------------------------
__global__ void gemm_scores(const bf16* __restrict__ qT, const bf16* __restrict__ kT,
                            float* __restrict__ escore, int mode) {
    const int b = blockIdx.y, f = blockIdx.x;
    const int wv = threadIdx.x >> 5, lane = threadIdx.x & 31;
    const bf16* qb = qT + (long)b * HW * CI;
    const bf16* kb = kT + (long)b * HW * CI;

    for (int st = wv; st < 49; st += 8) {
        int mt = st / 7, nt = st % 7;
        int m  = mt * 16 + (lane & 15); int mc = m < HH ? m : HH - 1;
        int n  = nt * 16 + (lane & 15); int nc = n < HH ? n : HH - 1;
        long apos = (mode == 0) ? ((long)mc * HH + f) : ((long)f * HH + mc);
        long bpos = (mode == 0) ? ((long)nc * HH + f) : ((long)f * HH + nc);
        const bf16* arow = qb + apos * CI;
        const bf16* bcol = kb + bpos * CI;

        v8f acc = {};
        #pragma unroll
        for (int k0 = 0; k0 < CI; k0 += 32) {
            v16bf a  = load_frag_a(arow, k0, lane);
            v16bf bb = load_frag_b(bcol, k0, lane);
            acc = wmma_bf16(a, bb, acc);
        }

        int mbase = mt * 16 + ((lane < 16) ? 0 : 8);
        int nout  = nt * 16 + (lane & 15);   // < 112, stays inside the 128-pad
        #pragma unroll
        for (int r = 0; r < 8; ++r) {
            int mm = mbase + r;
            if (mm < HH) {
                long row = (mode == 0) ? ((long)b * HW + (long)mm * HH + f)
                                       : ((long)b * HW + (long)f * HH + mm);
                escore[row * E2 + mode * KP + nout] = acc[r];
            }
        }
    }
}

// ---------------------------------------------------------------------------
// Fused masked softmax over concat [eH(97) | eW(97)], one wave32 per row.
// Writes zero-padded bf16 probs att (B,HW,256). grid: 9409 blocks x 256.
// ---------------------------------------------------------------------------
__global__ void softmax_rows(const float* __restrict__ escore, bf16* __restrict__ att) {
    const int wv = threadIdx.x >> 5, lane = threadIdx.x & 31;
    long row = (long)blockIdx.x * 8 + wv;
    if (row >= (long)BD * HW) return;
    int h = (int)(row % HW) / HH;
    const float* e = escore + row * E2;

    float vals[8];
    float mx = -3.4e38f;
    #pragma unroll
    for (int j = 0; j < 4; ++j) {
        int g = lane + 32 * j;
        float vH = (g < HH) ? (e[g] + ((g == h) ? NEGI : 0.0f)) : -3.4e38f;
        float vW = (g < HH) ? e[KP + g] : -3.4e38f;
        vals[j] = vH; vals[4 + j] = vW;
        mx = fmaxf(mx, fmaxf(vH, vW));
    }
    #pragma unroll
    for (int off = 16; off; off >>= 1) mx = fmaxf(mx, __shfl_xor(mx, off, 32));

    float sum = 0.0f;
    #pragma unroll
    for (int j = 0; j < 8; ++j) {
        float p = (vals[j] > -1.0e37f) ? __expf(vals[j] - mx) : 0.0f;
        vals[j] = p; sum += p;
    }
    #pragma unroll
    for (int off = 16; off; off >>= 1) sum += __shfl_xor(sum, off, 32);
    float inv = 1.0f / sum;

    #pragma unroll
    for (int j = 0; j < 4; ++j) {
        int g = lane + 32 * j;                      // covers 0..127
        float gate = (g < HH) ? 1.0f : 0.0f;
        att[row * E2 + g]      = (bf16)(vals[j]     * inv * gate);
        att[row * E2 + KP + g] = (bf16)(vals[4 + j] * inv * gate);
    }
}

// ---------------------------------------------------------------------------
// outH per (b,w): D[c,h] = sum_g vT[b,c,w,g] * aH[b,h,w,g].  K=128 (zero-pad).
// grid: (97, 8, B). Writes f32 scratch outHs (B,C,HW).
// ---------------------------------------------------------------------------
__global__ void gemm_outH(const bf16* __restrict__ vT, const bf16* __restrict__ att,
                          float* __restrict__ outHs) {
    const int w = blockIdx.x, mt = blockIdx.y, b = blockIdx.z;
    const int wv = threadIdx.x >> 5, lane = threadIdx.x & 31;
    const int msub = wv & 3, ngrp = wv >> 2;

    const int crow = mt * 64 + msub * 16 + (lane & 15);
    const bf16* arow = vT + (((long)b * CH + crow) * HH + w) * KP;
    const int cbase = mt * 64 + msub * 16 + ((lane < 16) ? 0 : 8);

    for (int tt = 0; tt < 4; ++tt) {
        int nt = ngrp + 2 * tt;
        if (nt >= 7) break;                       // uniform per wave
        int n  = nt * 16 + (lane & 15);
        int nc = n < HH ? n : HH - 1;
        const bf16* bcol = att + ((long)b * HW + (long)nc * HH + w) * E2;

        v8f acc = {};
        #pragma unroll
        for (int k0 = 0; k0 < KP; k0 += 32) {
            v16bf a  = load_frag_a(arow, k0, lane);
            v16bf bb = load_frag_b(bcol, k0, lane);
            acc = wmma_bf16(a, bb, acc);
        }
        if (n < HH) {
            #pragma unroll
            for (int r = 0; r < 8; ++r)
                outHs[((long)b * CH + cbase + r) * HW + (long)n * HH + w] = acc[r];
        }
    }
}

// ---------------------------------------------------------------------------
// outW per (b,h) + epilogue: out = gamma*(outW + outHs) + x.  grid: (97, 8, B).
// ---------------------------------------------------------------------------
__global__ void gemm_outW(const bf16* __restrict__ vW, const bf16* __restrict__ att,
                          const float* __restrict__ outHs, const float* __restrict__ xin,
                          const float* __restrict__ gamma, float* __restrict__ out) {
    const int h = blockIdx.x, mt = blockIdx.y, b = blockIdx.z;
    const int wv = threadIdx.x >> 5, lane = threadIdx.x & 31;
    const int msub = wv & 3, ngrp = wv >> 2;

    const int crow = mt * 64 + msub * 16 + (lane & 15);
    const bf16* arow = vW + (((long)b * CH + crow) * HH + h) * KP;
    const int cbase = mt * 64 + msub * 16 + ((lane < 16) ? 0 : 8);
    const float gm = *gamma;

    for (int tt = 0; tt < 4; ++tt) {
        int nt = ngrp + 2 * tt;
        if (nt >= 7) break;
        int n  = nt * 16 + (lane & 15);
        int nc = n < HH ? n : HH - 1;
        const bf16* bcol = att + ((long)b * HW + (long)h * HH + nc) * E2 + KP;

        v8f acc = {};
        #pragma unroll
        for (int k0 = 0; k0 < KP; k0 += 32) {
            v16bf a  = load_frag_a(arow, k0, lane);
            v16bf bb = load_frag_b(bcol, k0, lane);
            acc = wmma_bf16(a, bb, acc);
        }
        if (n < HH) {
            #pragma unroll
            for (int r = 0; r < 8; ++r) {
                long idx = ((long)b * CH + cbase + r) * HW + (long)h * HH + n;
                out[idx] = gm * (acc[r] + outHs[idx]) + xin[idx];
            }
        }
    }
}

// ---------------------------------------------------------------------------
extern "C" void kernel_launch(void* const* d_in, const int* in_sizes, int n_in,
                              void* d_out, int out_size, void* d_ws, size_t ws_size,
                              hipStream_t stream) {
    const float* x_in  = (const float*)d_in[0];
    const float* Wq    = (const float*)d_in[1];
    const float* bq    = (const float*)d_in[2];
    const float* Wk    = (const float*)d_in[3];
    const float* bk    = (const float*)d_in[4];
    const float* Wv    = (const float*)d_in[5];
    const float* bv    = (const float*)d_in[6];
    const float* gamma = (const float*)d_in[7];
    float* out = (float*)d_out;

    char* ws = (char*)d_ws;
    size_t o = 0;
    auto take = [&](size_t bytes) { size_t r = o; o += (bytes + 255) & ~size_t(255); return r; };
    bf16*  wq_bf  = (bf16*) (ws + take((size_t)CI * CH * 2));
    bf16*  wk_bf  = (bf16*) (ws + take((size_t)CI * CH * 2));
    bf16*  wv_bf  = (bf16*) (ws + take((size_t)CH * CH * 2));
    bf16*  xT     = (bf16*) (ws + take((size_t)BD * HW * CH * 2));
    bf16*  qT     = (bf16*) (ws + take((size_t)BD * HW * CI * 2));
    bf16*  kT     = (bf16*) (ws + take((size_t)BD * HW * CI * 2));
    bf16*  vT     = (bf16*) (ws + take((size_t)BD * CH * HH * KP * 2));
    bf16*  vW     = (bf16*) (ws + take((size_t)BD * CH * HH * KP * 2));
    float* escore = (float*)(ws + take((size_t)BD * HW * E2 * 4));
    bf16*  att    = (bf16*) (ws + take((size_t)BD * HW * E2 * 2));
    float* outHs  = (float*)(ws + take((size_t)BD * CH * HW * 4));
    (void)ws_size; (void)in_sizes; (void)n_in; (void)out_size;

    // one-time (per launch) weight casts + pad zeroing
    cvt_f32_bf16<<<(CI * CH + 255) / 256, 256, 0, stream>>>(Wq, wq_bf, CI * CH);
    cvt_f32_bf16<<<(CI * CH + 255) / 256, 256, 0, stream>>>(Wk, wk_bf, CI * CH);
    cvt_f32_bf16<<<(CH * CH + 255) / 256, 256, 0, stream>>>(Wv, wv_bf, CH * CH);
    {
        long padThreads = (long)BD * CH * HH * 32;
        pad_zero<<<(unsigned)((padThreads + 255) / 256), 256, 0, stream>>>(vT, vW);
    }

    const long xElems = (long)BD * CH * HW;
    for (int it = 0; it < 2; ++it) {
        const float* xs = (it == 0) ? x_in : out;

        pack_xT<<<(unsigned)((xElems + 255) / 256), 256, 0, stream>>>(xs, xT);

        gemm_qk<<<dim3(148, BD), 256, 0, stream>>>(wq_bf, bq, xT, qT);
        gemm_qk<<<dim3(148, BD), 256, 0, stream>>>(wk_bf, bk, xT, kT);
        gemm_v <<<dim3(148, 8, BD), 256, 0, stream>>>(wv_bf, bv, xT, vT, vW);

        gemm_scores<<<dim3(HH, BD), 256, 0, stream>>>(qT, kT, escore, 0);
        gemm_scores<<<dim3(HH, BD), 256, 0, stream>>>(qT, kT, escore, 1);

        softmax_rows<<<(unsigned)HW, 256, 0, stream>>>(escore, att);

        gemm_outH<<<dim3(HH, 8, BD), 256, 0, stream>>>(vT, att, outHs);
        gemm_outW<<<dim3(HH, 8, BD), 256, 0, stream>>>(vW, att, outHs, xs, gamma, out);
    }
}
